// GMVAE_19877108646294
// MI455X (gfx1250) — matrix operand order repaired and checked by reading, not compiled, not run on hardware
//
#include <hip/hip_runtime.h>
#include <hip/hip_bf16.h>
#include <math.h>

typedef __attribute__((ext_vector_type(16))) _Float16 v16h;
typedef __attribute__((ext_vector_type(8)))  _Float16 v8h;
typedef __attribute__((ext_vector_type(8)))  float    v8f;

#define NB    8       // B
#define NS    32      // S
#define NMC   32      // MC
#define NHID  64      // HID
#define NLAT  64      // LAT
#define NK    16      // K (mixture comps)
#define NIMG  256     // B*S
#define NLH   256     // 4*HID
#define NZ    8192    // B*S*MC
#define NPC   1024    // S*MC
#define LOGC_ (-0.918938533204672742f)
#define BNEPS 1e-5f

// -------- CDNA5 async global->LDS copy (ASYNCcnt path), 16B per lane --------
__device__ __forceinline__ void async_g2l_b128(void* lds, const void* gaddr) {
    unsigned l = (unsigned)(unsigned long long)lds;   // low 32 bits = LDS byte offset
    asm volatile("global_load_async_to_lds_b128 %0, %1, off"
                 :: "v"(l), "v"(gaddr)
                 : "memory");
}
__device__ __forceinline__ void wait_async0() {
    asm volatile("s_wait_asynccnt 0" ::: "memory");
}

// ---------------------------------------------------------------- utilities
__global__ void k_cast_f16(const float* __restrict__ in, _Float16* __restrict__ out, int n) {
    int i = blockIdx.x * 256 + threadIdx.x;
    if (i < n) out[i] = (_Float16)in[i];
}
__global__ void k_zero(float* p, int n) {
    int i = blockIdx.x * 256 + threadIdx.x;
    if (i < n) p[i] = 0.0f;
}
__global__ void k_copy(const float* __restrict__ s, float* __restrict__ d, int n) {
    int i = blockIdx.x * 256 + threadIdx.x;
    if (i < n) d[i] = s[i];
}

// Pack a row-major f32 K x N weight matrix into f16 WMMA B-fragment order:
// out[((kblk*ntile + tn)*512) + lane*16 + e] = B[(kblk*32 + (lane>=16)*16 + e)*N + tn*16 + lane%16]
__global__ void k_pack_b(const float* __restrict__ B, _Float16* __restrict__ out, int K, int N) {
    int idx = blockIdx.x * 256 + threadIdx.x;
    if (idx >= K * N) return;
    int e    = idx & 15;
    int lane = (idx >> 4) & 31;
    int tile = idx >> 9;
    int ntile = N >> 4;
    int tn   = tile % ntile;
    int kblk = tile / ntile;
    int k   = kblk * 32 + (lane >> 4) * 16 + e;
    int col = tn * 16 + (lane & 15);
    out[idx] = (_Float16)B[(long)k * N + col];
}

// ---------------------------------------------------------------- WMMA GEMM
// C(MxN) = act(A(MxK) @ B + bias) [+ residual]; A f16 row-major, B pre-packed
// fragment-order f16; C f32 (+ optional f16 copy). One wave computes TWO
// adjacent 16x16 N-tiles sharing one A fragment; K loop fully unrolled.
template <int KD, int ACT, bool RESID, bool WRH>
__global__ void k_gemm2(const _Float16* __restrict__ A, const _Float16* __restrict__ Bp,
                        const float* __restrict__ bias, const float* __restrict__ res,
                        float* __restrict__ C, _Float16* __restrict__ Ch, int M, int N) {
    int lane = threadIdx.x & 31;
    int tile = blockIdx.x * 4 + (threadIdx.x >> 5);
    int ng = N >> 5;                    // 32-column groups
    int tm = tile / ng, tg = tile % ng;
    int ntile = N >> 4;
    int row = tm * 16 + (lane & 15);
    int hi  = lane >> 4;
    const _Float16* Arow = A + (long)row * KD;

    v8f acc[2]; acc[0] = (v8f){}; acc[1] = (v8f){};
#pragma unroll
    for (int kb = 0; kb < KD / 32; ++kb) {
        int k0 = kb * 32;
        // A fragment: two contiguous 16B chunks per lane
        v8h a0 = *(const v8h*)(Arow + k0 + hi * 16);
        v8h a1 = *(const v8h*)(Arow + k0 + hi * 16 + 8);
        v16h a = __builtin_shufflevector(a0, a1, 0, 1, 2, 3, 4, 5, 6, 7,
                                         8, 9, 10, 11, 12, 13, 14, 15);
        // B fragments: packed, 32B contiguous per lane per tile
        const _Float16* bb = Bp + ((long)(kb * ntile + tg * 2) << 9) + lane * 16;
        v8h b0l = *(const v8h*)(bb);
        v8h b0h = *(const v8h*)(bb + 8);
        v8h b1l = *(const v8h*)(bb + 512);
        v8h b1h = *(const v8h*)(bb + 520);
        v16h b0 = __builtin_shufflevector(b0l, b0h, 0, 1, 2, 3, 4, 5, 6, 7,
                                          8, 9, 10, 11, 12, 13, 14, 15);
        v16h b1 = __builtin_shufflevector(b1l, b1h, 0, 1, 2, 3, 4, 5, 6, 7,
                                          8, 9, 10, 11, 12, 13, 14, 15);
        acc[0] = __builtin_amdgcn_wmma_f32_16x16x32_f16(false, a, false, b0, (short)0, acc[0], false, false);
        acc[1] = __builtin_amdgcn_wmma_f32_16x16x32_f16(false, a, false, b1, (short)0, acc[1], false, false);
    }
#pragma unroll
    for (int t = 0; t < 2; ++t) {
        int col = tg * 32 + t * 16 + (lane & 15);
        float bsv = bias[col];
#pragma unroll
        for (int r = 0; r < 8; ++r) {
            int orow = tm * 16 + r + hi * 8;
            float v = acc[t][r] + bsv;
            if (ACT == 1)      v = v > 0.0f ? v : 0.0f;
            else if (ACT == 2) v = v > 0.0f ? v : (__expf(v) - 1.0f);
            if (RESID) v += res[(long)orow * N + col];
            C[(long)orow * N + col] = v;
            if (WRH) Ch[(long)orow * N + col] = (_Float16)v;
        }
    }
}

// ---------------------------------------------------------------- encoder
__global__ void k_conv3x3(const float* __restrict__ x, const float* __restrict__ w,
                          const float* __restrict__ b, float* __restrict__ y,
                          int nimg, int cin, int cout, int h) {
    int idx = blockIdx.x * 256 + threadIdx.x;
    int total = nimg * cout * h * h;
    if (idx >= total) return;
    int xw = idx % h; int t = idx / h; int yy = t % h; t /= h;
    int co = t % cout; int img = t / cout;
    float acc = b[co];
    for (int ci = 0; ci < cin; ++ci) {
        const float* xb = x + ((long)(img * cin + ci) * h) * h;
        const float* wb = w + ((long)(co * cin + ci) * 9);
        for (int ky = 0; ky < 3; ++ky) {
            int iy = yy + ky - 1; if (iy < 0 || iy >= h) continue;
            for (int kx = 0; kx < 3; ++kx) {
                int ix = xw + kx - 1; if (ix < 0 || ix >= h) continue;
                acc += xb[iy * h + ix] * wb[ky * 3 + kx];
            }
        }
    }
    y[idx] = acc;
}

__global__ void k_bn_stats(const float* __restrict__ x, float* __restrict__ stats,
                           int nimg, int C, int sp) {
    __shared__ float ls[256], lq[256];
    int c = blockIdx.x, tid = threadIdx.x;
    int total = nimg * sp;
    float s = 0.0f, q = 0.0f;
    for (int i = tid; i < total; i += 256) {
        int img = i / sp, k = i - img * sp;
        float v = x[((long)(img * C + c)) * sp + k];
        s += v; q += v * v;
    }
    ls[tid] = s; lq[tid] = q; __syncthreads();
    for (int o = 128; o > 0; o >>= 1) {
        if (tid < o) { ls[tid] += ls[tid + o]; lq[tid] += lq[tid + o]; }
        __syncthreads();
    }
    if (tid == 0) {
        float m = ls[0] / (float)total;
        stats[c] = m;
        stats[C + c] = lq[0] / (float)total - m * m;
    }
}

__global__ void k_bn_relu_pool(const float* __restrict__ x, const float* __restrict__ st,
                               const float* __restrict__ g, const float* __restrict__ be,
                               float* __restrict__ y, int nimg, int C, int h) {
    int hp = h >> 1;
    int idx = blockIdx.x * 256 + threadIdx.x;
    int total = nimg * C * hp * hp;
    if (idx >= total) return;
    int px = idx % hp; int t = idx / hp; int py = t % hp; t /= hp;
    int c = t % C; int img = t / C;
    float m = st[c], rs = rsqrtf(st[C + c] + BNEPS), gg = g[c], bb = be[c];
    const float* base = x + ((long)(img * C + c) * h + py * 2) * h + px * 2;
    float best = 0.0f;  // post-relu values are >= 0
    for (int dy = 0; dy < 2; ++dy)
        for (int dx = 0; dx < 2; ++dx) {
            float v = gg * (base[dy * h + dx] - m) * rs + bb;
            v = v > 0.0f ? v : 0.0f;
            best = fmaxf(best, v);
        }
    y[idx] = best;
}

// ---------------------------------------------------------------- attention core
__global__ void k_attn(const float* __restrict__ Q, const float* __restrict__ Kt,
                       const float* __restrict__ V, float* __restrict__ O,
                       _Float16* __restrict__ Oh) {
    int bh = blockIdx.x; int b = bh >> 2, h = bh & 3;
    int q = threadIdx.x;                 // 32 threads, 1 wave
    int rb = b * 32, coff = h * 64;
    float s[32]; float mx = -1e30f;
    for (int k = 0; k < 32; ++k) {
        float dot = 0.0f;
        for (int d = 0; d < 64; ++d)
            dot += Q[(long)(rb + q) * NLH + coff + d] * Kt[(long)(rb + k) * NLH + coff + d];
        s[k] = dot * 0.0625f;            // 1/sqrt(256)
        mx = fmaxf(mx, s[k]);
    }
    float sum = 0.0f;
    for (int k = 0; k < 32; ++k) { s[k] = __expf(s[k] - mx); sum += s[k]; }
    float inv = 1.0f / sum;
    for (int d = 0; d < 64; ++d) {
        float acc = 0.0f;
        for (int k = 0; k < 32; ++k) acc += s[k] * V[(long)(rb + k) * NLH + coff + d];
        float o = Q[(long)(rb + q) * NLH + coff + d] + acc * inv;
        O[(long)(rb + q) * NLH + coff + d] = o;
        Oh[(long)(rb + q) * NLH + coff + d] = (_Float16)o;
    }
}

// ---------------------------------------------------------------- reparameterize
__global__ void k_z(const float* __restrict__ ml, const float* __restrict__ eps,
                    float* __restrict__ z, _Float16* __restrict__ zh) {
    int idx = blockIdx.x * 256 + threadIdx.x;
    if (idx >= NIMG * NMC * NLAT) return;
    int d = idx % NLAT; int t = idx / NLAT; int mc = t % NMC; int bs = t / NMC;
    float m  = ml[bs * 128 + d];
    float lv = ml[bs * 128 + 64 + d];
    float val = m + __expf(0.5f * lv) * eps[(long)(bs * NMC + mc) * NLAT + d];
    int b = bs >> 5, s = bs & 31; int n = s * 32 + mc;
    long o = ((long)b * NPC + n) * NLAT + d;
    z[o] = val; zh[o] = (_Float16)val;
}

// ---------------------------------------------------------------- EM prior
__global__ void k_em_init(const float* __restrict__ z, const int* __restrict__ idx,
                          float* __restrict__ mean, float* __restrict__ pi) {
    int i = blockIdx.x * 256 + threadIdx.x;
    if (i >= NB * NK * NLAT) return;
    int d = i % NLAT; int t = i / NLAT; int k = t % NK; int b = t / NK;
    int n = idx[b * NK + k];
    mean[i] = z[((long)b * NPC + n) * NLAT + d];
    if (d == 0) pi[b * NK + k] = 1.0f / NK;
}

__global__ void k_em_post(const float* __restrict__ z, const float* __restrict__ pi,
                          const float* __restrict__ mean, float* __restrict__ post) {
    int i = blockIdx.x * 256 + threadIdx.x;
    if (i >= NB * NPC) return;
    int n = i % NPC, b = i / NPC;
    const float* zp = z + ((long)b * NPC + n) * NLAT;
    float ll[NK]; float mx = -1e30f;
    for (int k = 0; k < NK; ++k) {
        const float* mp = mean + (long)(b * NK + k) * NLAT;
        float d2 = 0.0f;
        for (int d = 0; d < NLAT; ++d) { float df = zp[d] - mp[d]; d2 += df * df; }
        float l = -0.5f * d2 + NLAT * LOGC_ + __logf(pi[b * NK + k]);
        ll[k] = l; mx = fmaxf(mx, l);
    }
    float s = 0.0f;
    for (int k = 0; k < NK; ++k) { ll[k] = __expf(ll[k] - mx); s += ll[k]; }
    float inv = 1.0f / s;
    for (int k = 0; k < NK; ++k) post[(long)i * NK + k] = ll[k] * inv;
}

__global__ void k_em_nk(const float* __restrict__ post, float* __restrict__ Nk,
                        float* __restrict__ piOut) {
    __shared__ float ls[256];
    int bk = blockIdx.x; int b = bk / NK, k = bk % NK;
    float s = 0.0f;
    for (int n = threadIdx.x; n < NPC; n += 256)
        s += post[((long)b * NPC + n) * NK + k];
    ls[threadIdx.x] = s; __syncthreads();
    for (int o = 128; o > 0; o >>= 1) {
        if (threadIdx.x < o) ls[threadIdx.x] += ls[threadIdx.x + o];
        __syncthreads();
    }
    if (threadIdx.x == 0) { Nk[bk] = ls[0]; piOut[bk] = ls[0] * (1.0f / NPC); }
}

__global__ void k_em_mean(const float* __restrict__ post, const float* __restrict__ z,
                          const float* __restrict__ Nk, float* __restrict__ meanOut) {
    int i = blockIdx.x * 256 + threadIdx.x;
    if (i >= NB * NK * NLAT) return;
    int d = i % NLAT; int t = i / NLAT; int k = t % NK; int b = t / NK;
    float s = 0.0f;
    for (int n = 0; n < NPC; ++n)
        s += post[((long)b * NPC + n) * NK + k] * z[((long)b * NPC + n) * NLAT + d];
    meanOut[i] = s / Nk[b * NK + k];
}

// ---------------------------------------------------------------- decoder convT
// transposed conv, stride 2, pad 1; w layout (Cin, Cout, k, k)
__global__ void k_convT(const float* __restrict__ x, const float* __restrict__ w,
                        const float* __restrict__ bias, float* __restrict__ y,
                        int nimg, int cin, int cout, int ih, int oh, int kk) {
    long idx = (long)blockIdx.x * 256 + threadIdx.x;
    long total = (long)nimg * cout * oh * oh;
    if (idx >= total) return;
    int ox = idx % oh; long t = idx / oh; int oy = t % oh; t /= oh;
    int co = t % cout; int img = t / cout;
    float acc = bias[co];
    const float* xb = x + (long)img * cin * ih * ih;
    for (int ky = 0; ky < kk; ++ky) {
        int ty = oy + 1 - ky; if (ty < 0 || (ty & 1)) continue;
        int iy = ty >> 1; if (iy >= ih) continue;
        for (int kx = 0; kx < kk; ++kx) {
            int tx = ox + 1 - kx; if (tx < 0 || (tx & 1)) continue;
            int ix = tx >> 1; if (ix >= ih) continue;
            for (int ci = 0; ci < cin; ++ci)
                acc += xb[(ci * ih + iy) * ih + ix] * w[((long)(ci * cout + co)) * kk * kk + ky * kk + kx];
        }
    }
    y[idx] = acc;
}

__global__ void k_bn_relu(float* __restrict__ x, const float* __restrict__ st,
                          const float* __restrict__ g, const float* __restrict__ be,
                          int nimg, int C, int sp) {
    long idx = (long)blockIdx.x * 256 + threadIdx.x;
    long total = (long)nimg * C * sp;
    if (idx >= total) return;
    int c = (int)((idx / sp) % C);
    float v = g[c] * (x[idx] - st[c]) * rsqrtf(st[C + c] + BNEPS) + be[c];
    x[idx] = v > 0.0f ? v : 0.0f;
}

// streaming BN-stats for convT2 (never materialize the 411MB tensor);
// x2 tile staged into LDS via CDNA5 async global->LDS copies.
__global__ void k_ct2_stats(const float* __restrict__ x2, const float* __restrict__ w2,
                            const float* __restrict__ b2, float* __restrict__ gsum,
                            float* __restrict__ gsq) {
    __shared__ float xs[64 * 49];
    __shared__ float ssum[64], ssq[64];
    int img = blockIdx.x, tid = threadIdx.x;
    for (int i = tid * 4; i < 64 * 49; i += 1024)
        async_g2l_b128(&xs[i], x2 + (long)img * 64 * 49 + i);
    if (tid < 64) { ssum[tid] = 0.0f; ssq[tid] = 0.0f; }
    wait_async0();
    __syncthreads();
    for (int o = tid; o < 64 * 196; o += 256) {
        int co = o / 196; int sp = o % 196; int oy = sp / 14, ox = sp % 14;
        float acc = b2[co];
        for (int ky = 0; ky < 4; ++ky) {
            int ty = oy + 1 - ky; if (ty < 0 || (ty & 1)) continue;
            int iy = ty >> 1; if (iy >= 7) continue;
            for (int kx = 0; kx < 4; ++kx) {
                int tx = ox + 1 - kx; if (tx < 0 || (tx & 1)) continue;
                int ix = tx >> 1; if (ix >= 7) continue;
                for (int ci = 0; ci < 64; ++ci)
                    acc += xs[ci * 49 + iy * 7 + ix] * w2[(ci * 64 + co) * 16 + ky * 4 + kx];
            }
        }
        atomicAdd(&ssum[co], acc);
        atomicAdd(&ssq[co], acc * acc);
    }
    __syncthreads();
    if (tid < 64) { atomicAdd(&gsum[tid], ssum[tid]); atomicAdd(&gsq[tid], ssq[tid]); }
}

__global__ void k_ct2_finalize(const float* __restrict__ gsum, const float* __restrict__ gsq,
                               float* __restrict__ st2) {
    int c = threadIdx.x;  // 64 threads
    float cnt = 8192.0f * 196.0f;
    float m = gsum[c] / cnt;
    st2[c] = m;
    st2[64 + c] = gsq[c] / cnt - m * m;
}

// fused: convT2 -> BN -> relu (LDS-resident 64x14x14 tile) -> convT3 -> sigmoid -> out
__global__ void k_final(const float* __restrict__ x2, const float* __restrict__ w2,
                        const float* __restrict__ b2, const float* __restrict__ st2,
                        const float* __restrict__ g2, const float* __restrict__ be2,
                        const float* __restrict__ w3, const float* __restrict__ b3,
                        float* __restrict__ out) {
    __shared__ float xs[64 * 49];
    __shared__ float x3[64 * 196];
    int img = blockIdx.x, tid = threadIdx.x;
    for (int i = tid * 4; i < 64 * 49; i += 1024)
        async_g2l_b128(&xs[i], x2 + (long)img * 64 * 49 + i);
    wait_async0();
    __syncthreads();
    for (int o = tid; o < 64 * 196; o += 256) {
        int co = o / 196; int sp = o % 196; int oy = sp / 14, ox = sp % 14;
        float acc = b2[co];
        for (int ky = 0; ky < 4; ++ky) {
            int ty = oy + 1 - ky; if (ty < 0 || (ty & 1)) continue;
            int iy = ty >> 1; if (iy >= 7) continue;
            for (int kx = 0; kx < 4; ++kx) {
                int tx = ox + 1 - kx; if (tx < 0 || (tx & 1)) continue;
                int ix = tx >> 1; if (ix >= 7) continue;
                for (int ci = 0; ci < 64; ++ci)
                    acc += xs[ci * 49 + iy * 7 + ix] * w2[(ci * 64 + co) * 16 + ky * 4 + kx];
            }
        }
        float v = g2[co] * (acc - st2[co]) * rsqrtf(st2[64 + co] + BNEPS) + be2[co];
        x3[o] = v > 0.0f ? v : 0.0f;
    }
    __syncthreads();
    for (int o = tid; o < 784; o += 256) {
        int oy = o / 28, ox = o % 28;
        float acc = b3[0];
        for (int ky = 0; ky < 4; ++ky) {
            int ty = oy + 1 - ky; if (ty < 0 || (ty & 1)) continue;
            int iy = ty >> 1; if (iy >= 14) continue;
            for (int kx = 0; kx < 4; ++kx) {
                int tx = ox + 1 - kx; if (tx < 0 || (tx & 1)) continue;
                int ix = tx >> 1; if (ix >= 14) continue;
                for (int ci = 0; ci < 64; ++ci)
                    acc += x3[ci * 196 + iy * 14 + ix] * w3[ci * 16 + ky * 4 + kx];
            }
        }
        out[(long)img * 784 + o] = 1.0f / (1.0f + __expf(-acc));
    }
}

// ---------------------------------------------------------------- host
static inline dim3 g1(long n) { return dim3((unsigned)((n + 255) / 256)); }
static inline dim3 gt(int M, int N) { return dim3((unsigned)(((M / 16) * (N / 32)) / 4)); }

extern "C" void kernel_launch(void* const* d_in, const int* in_sizes, int n_in,
                              void* d_out, int out_size, void* d_ws, size_t ws_size,
                              hipStream_t stream) {
    (void)in_sizes; (void)n_in; (void)out_size; (void)ws_size;
    // ---- inputs (setup_inputs() dict order, nested dicts flattened in insertion order)
    const float* X       = (const float*)d_in[0];
    const float* eps     = (const float*)d_in[1];
    const int*   initIdx = (const int*)d_in[2];
    int p = 3;
    const float *ew[4], *eb[4], *eg[4], *ebe[4];
    for (int i = 0; i < 4; ++i) {
        ew[i]  = (const float*)d_in[p++]; eb[i]  = (const float*)d_in[p++];
        eg[i]  = (const float*)d_in[p++]; ebe[i] = (const float*)d_in[p++];
    }
    const float *sWq[2], *sbq[2], *sWk[2], *sbk[2], *sWv[2], *sbv[2], *sWo[2], *sbo[2];
    for (int i = 0; i < 2; ++i) {
        sWq[i] = (const float*)d_in[p++]; sbq[i] = (const float*)d_in[p++];
        sWk[i] = (const float*)d_in[p++]; sbk[i] = (const float*)d_in[p++];
        sWv[i] = (const float*)d_in[p++]; sbv[i] = (const float*)d_in[p++];
        sWo[i] = (const float*)d_in[p++]; sbo[i] = (const float*)d_in[p++];
    }
    const float* Wl = (const float*)d_in[p++];
    const float* bl = (const float*)d_in[p++];
    const float *pW[3], *pb[3];
    for (int i = 0; i < 3; ++i) { pW[i] = (const float*)d_in[p++]; pb[i] = (const float*)d_in[p++]; }
    const float *dw[4], *db[4], *dg[3], *dbe[3];
    for (int i = 0; i < 4; ++i) {
        dw[i] = (const float*)d_in[p++]; db[i] = (const float*)d_in[p++];
        if (i < 3) { dg[i] = (const float*)d_in[p++]; dbe[i] = (const float*)d_in[p++]; }
    }

    // ---- workspace layout (floats)
    float* fw = (float*)d_ws;
    float* A0 = fw;                       // 26,214,400
    float* A1 = fw + 26214400;            // 26,214,400
    float* SC = fw + 52428800;            //  4,194,304
    float *Qb = SC, *Kb = SC + 65536, *Vb = SC + 131072, *Ob = SC + 196608;
    float* ml   = SC + 262144;            // 32768
    float* z    = SC + 294912;            // 524288
    float* post = SC + 819200;            // 131072
    float* Nk   = SC + 950272;            // 128
    float* piA  = SC + 950400;  float* piB  = SC + 950528;
    float* mnA  = SC + 950656;  float* mnB  = SC + 958848;     // 8192 each
    float* stE  = SC + 967040;            // 128 (encoder/decoder BN stats, reused)
    float* gsum = SC + 967168;  float* gsq = SC + 967232;      // 64 + 64
    float* st2  = SC + 967296;            // 128
    _Float16* HB = (_Float16*)(fw + 56623104);
    _Float16* h0 = HB;                    // 2,097,152 halves
    _Float16* h1 = HB + 2097152;
    _Float16* hw = HB + 4194304;          // packed weight halves + zh
    _Float16 *hWq[2], *hWk[2], *hWv[2], *hWo[2];
    for (int i = 0; i < 2; ++i) {
        hWq[i] = hw + i * 262144;          hWk[i] = hw + i * 262144 + 65536;
        hWv[i] = hw + i * 262144 + 131072; hWo[i] = hw + i * 262144 + 196608;
    }
    _Float16* hWl = hw + 524288;
    _Float16* hW1 = hw + 557056;
    _Float16* hW2 = hw + 573440;
    _Float16* hW3 = hw + 638976;
    _Float16* zh  = hw + 786432;          // 524288 halves

    // ---- pack weights used by WMMA GEMMs into f16 fragment order
    for (int i = 0; i < 2; ++i) {
        k_pack_b<<<g1(65536), 256, 0, stream>>>(sWq[i], hWq[i], 256, 256);
        k_pack_b<<<g1(65536), 256, 0, stream>>>(sWk[i], hWk[i], 256, 256);
        k_pack_b<<<g1(65536), 256, 0, stream>>>(sWv[i], hWv[i], 256, 256);
        k_pack_b<<<g1(65536), 256, 0, stream>>>(sWo[i], hWo[i], 256, 256);
    }
    k_pack_b<<<g1(32768), 256, 0, stream>>>(Wl, hWl, 256, 128);
    k_pack_b<<<g1(16384), 256, 0, stream>>>(pW[0], hW1, 64, 256);
    k_pack_b<<<g1(65536), 256, 0, stream>>>(pW[1], hW2, 256, 256);
    k_pack_b<<<g1(65536), 256, 0, stream>>>(pW[2], hW3, 256, 256);

    // ---- encoder: 4x [conv3x3 -> BN -> relu -> maxpool2]
    k_conv3x3<<<g1((long)NIMG * 64 * 32 * 32), 256, 0, stream>>>(X, ew[0], eb[0], A0, NIMG, 1, 64, 32);
    k_bn_stats<<<64, 256, 0, stream>>>(A0, stE, NIMG, 64, 32 * 32);
    k_bn_relu_pool<<<g1((long)NIMG * 64 * 16 * 16), 256, 0, stream>>>(A0, stE, eg[0], ebe[0], A1, NIMG, 64, 32);
    int hcur = 16;
    for (int L = 1; L < 4; ++L) {
        k_conv3x3<<<g1((long)NIMG * 64 * hcur * hcur), 256, 0, stream>>>(A1, ew[L], eb[L], A0, NIMG, 64, 64, hcur);
        k_bn_stats<<<64, 256, 0, stream>>>(A0, stE, NIMG, 64, hcur * hcur);
        int hp = hcur >> 1;
        k_bn_relu_pool<<<g1((long)NIMG * 64 * hp * hp), 256, 0, stream>>>(A0, stE, eg[L], ebe[L], A1, NIMG, 64, hcur);
        hcur = hp;
    }
    // A1 now holds H: (256 x 256), NCHW flatten == reshape(B,S,LH)

    // ---- two SAB blocks (all projections via WMMA)
    k_cast_f16<<<g1(65536), 256, 0, stream>>>(A1, h0, 65536);
    for (int i = 0; i < 2; ++i) {
        k_gemm2<256, 0, false, false><<<gt(NIMG, NLH), 128, 0, stream>>>(h0, hWq[i], sbq[i], nullptr, Qb, nullptr, NIMG, NLH);
        k_gemm2<256, 0, false, false><<<gt(NIMG, NLH), 128, 0, stream>>>(h0, hWk[i], sbk[i], nullptr, Kb, nullptr, NIMG, NLH);
        k_gemm2<256, 0, false, false><<<gt(NIMG, NLH), 128, 0, stream>>>(h0, hWv[i], sbv[i], nullptr, Vb, nullptr, NIMG, NLH);
        k_attn<<<32, 32, 0, stream>>>(Qb, Kb, Vb, Ob, h1);
        // X_out = O + relu(O @ Wo + bo); f32 dump to A0, f16 to h0 for next stage
        k_gemm2<256, 1, true, true><<<gt(NIMG, NLH), 128, 0, stream>>>(h1, hWo[i], sbo[i], Ob, A0, h0, NIMG, NLH);
    }
    // ml = Hq @ Wl + bl  (256 x 128)
    k_gemm2<256, 0, false, false><<<gt(NIMG, 128), 128, 0, stream>>>(h0, hWl, bl, nullptr, ml, nullptr, NIMG, 128);

    // ---- reparameterize z
    k_z<<<g1((long)NIMG * NMC * NLAT), 256, 0, stream>>>(ml, eps, z, zh);

    // ---- EM prior (5 iterations, ping-pong)
    k_em_init<<<g1(NB * NK * NLAT), 256, 0, stream>>>(z, initIdx, mnA, piA);
    float *piC = piA, *piN = piB, *mnC = mnA, *mnN = mnB;
    for (int it = 0; it < 5; ++it) {
        k_em_post<<<g1(NB * NPC), 256, 0, stream>>>(z, piC, mnC, post);
        k_em_nk<<<NB * NK, 256, 0, stream>>>(post, Nk, piN);
        k_em_mean<<<g1(NB * NK * NLAT), 256, 0, stream>>>(post, z, Nk, mnN);
        float* t;
        t = piC; piC = piN; piN = t;
        t = mnC; mnC = mnN; mnN = t;
    }

    // ---- decoder MLP (WMMA, elu)
    k_gemm2< 64, 2, false, true ><<<gt(NZ, NLH), 128, 0, stream>>>(zh, hW1, pb[0], nullptr, A0, h0, NZ, NLH);
    k_gemm2<256, 2, false, true ><<<gt(NZ, NLH), 128, 0, stream>>>(h0, hW2, pb[1], nullptr, A1, h1, NZ, NLH);
    k_gemm2<256, 2, false, false><<<gt(NZ, NLH), 128, 0, stream>>>(h1, hW3, pb[2], nullptr, A0, nullptr, NZ, NLH);
    // A0: (8192, 64, 2, 2)

    // ---- convT0 (k=4): 2->4, BN, relu
    k_convT<<<g1((long)NZ * 64 * 16), 256, 0, stream>>>(A0, dw[0], db[0], A1, NZ, 64, 64, 2, 4, 4);
    k_bn_stats<<<64, 256, 0, stream>>>(A1, stE, NZ, 64, 16);
    k_bn_relu<<<g1((long)NZ * 64 * 16), 256, 0, stream>>>(A1, stE, dg[0], dbe[0], NZ, 64, 16);
    // ---- convT1 (k=3): 4->7, BN, relu
    k_convT<<<g1((long)NZ * 64 * 49), 256, 0, stream>>>(A1, dw[1], db[1], A0, NZ, 64, 64, 4, 7, 3);
    k_bn_stats<<<64, 256, 0, stream>>>(A0, stE, NZ, 64, 49);
    k_bn_relu<<<g1((long)NZ * 64 * 49), 256, 0, stream>>>(A0, stE, dg[1], dbe[1], NZ, 64, 49);
    // ---- convT2 BN stats: streaming (no 411MB intermediate)
    k_zero<<<1, 128, 0, stream>>>(gsum, 128);  // gsum+gsq contiguous
    k_ct2_stats<<<NZ, 256, 0, stream>>>(A0, dw[2], db[2], gsum, gsq);
    k_ct2_finalize<<<1, 64, 0, stream>>>(gsum, gsq, st2);
    // ---- fused convT2 -> BN -> relu -> convT3 -> sigmoid -> x_rec
    k_final<<<NZ, 256, 0, stream>>>(A0, dw[2], db[2], st2, dg[2], dbe[2], dw[3], db[3], (float*)d_out);

    // ---- pi, mean outputs (after x_rec: 8192*784 floats)
    const long XRE = (long)NZ * 784;
    k_copy<<<1, 128, 0, stream>>>(piC, (float*)d_out + XRE, 128);
    k_copy<<<g1(NB * NK * NLAT), 256, 0, stream>>>(mnC, (float*)d_out + XRE + 128, NB * NK * NLAT);
}
